// MambaLM_58746562674796
// MI455X (gfx1250) — matrix operand order repaired and checked by reading, not compiled, not run on hardware
//
#include <hip/hip_runtime.h>
#include <hip/hip_bf16.h>
#include <math.h>

#define D_MODEL   768
#define N_LAYER   4
#define D_STATE   64
#define D_CONV    4
#define HEADDIM   64
#define D_INNER   1536
#define NHEADS    24
#define CONV_DIM  1664          // D_INNER + 2*D_STATE
#define D_IN_PROJ 3224          // 2*D_INNER + 2*D_STATE + NHEADS
#define VOCAB     50257
#define SEQ_L     1024
#define EPS_F     1e-5f

typedef __bf16 bf16_t;
typedef __attribute__((ext_vector_type(16))) __bf16 bf16x16;
typedef __attribute__((ext_vector_type(8)))  __bf16 bf16x8;
typedef __attribute__((ext_vector_type(8)))  float  f32x8;

// ---------------------------------------------------------------------------
// f32 -> bf16 conversion (grid-stride)
// ---------------------------------------------------------------------------
__global__ void f32_to_bf16_kernel(const float* __restrict__ src,
                                   bf16_t* __restrict__ dst, long n) {
  long i = (long)blockIdx.x * blockDim.x + threadIdx.x;
  long stride = (long)gridDim.x * blockDim.x;
  for (; i < n; i += stride) dst[i] = (__bf16)src[i];
}

// ---------------------------------------------------------------------------
// Embedding gather
// ---------------------------------------------------------------------------
__global__ void embed_kernel(const int* __restrict__ idx,
                             const float* __restrict__ emb,
                             float* __restrict__ x) {
  int t = blockIdx.x;
  long row = (long)idx[t] * D_MODEL;
  for (int d = threadIdx.x; d < D_MODEL; d += blockDim.x)
    x[(long)t * D_MODEL + d] = emb[row + d];
}

// ---------------------------------------------------------------------------
// LayerNorm (f32 in) -> bf16 out. One block per row.
// ---------------------------------------------------------------------------
__global__ __launch_bounds__(256)
void layernorm_bf16_kernel(const float* __restrict__ x,
                           const float* __restrict__ w,
                           const float* __restrict__ b,
                           bf16_t* __restrict__ out, int D) {
  __shared__ float red[256];
  int t = blockIdx.x;
  const float* row = x + (long)t * D;
  float s = 0.f;
  for (int d = threadIdx.x; d < D; d += 256) s += row[d];
  red[threadIdx.x] = s; __syncthreads();
  for (int o = 128; o > 0; o >>= 1) {
    if (threadIdx.x < o) red[threadIdx.x] += red[threadIdx.x + o];
    __syncthreads();
  }
  float mean = red[0] / (float)D;
  __syncthreads();
  float v = 0.f;
  for (int d = threadIdx.x; d < D; d += 256) {
    float dd = row[d] - mean; v += dd * dd;
  }
  red[threadIdx.x] = v; __syncthreads();
  for (int o = 128; o > 0; o >>= 1) {
    if (threadIdx.x < o) red[threadIdx.x] += red[threadIdx.x + o];
    __syncthreads();
  }
  float rstd = rsqrtf(red[0] / (float)D + EPS_F);
  for (int d = threadIdx.x; d < D; d += 256)
    out[(long)t * D + d] = (__bf16)((row[d] - mean) * rstd * w[d] + b[d]);
}

// ---------------------------------------------------------------------------
// Fragment helpers (CDNA5 ISA 7.12.2, wave32):
//   A 16x32 bf16: lane<16 -> K in {k0..+7, k0+16..+23}; lane>=16 -> +8
//   B 32x16 bf16: N = lane&15; K split as for A: {k0..+7, k0+16..+23}; lane>=16 -> +8
//   C/D f32     : N = lane&15; vgpr r -> M = r + 8*(lane>=16)
// ---------------------------------------------------------------------------
__device__ __forceinline__ bf16x16 load_a_frag(const bf16_t* Arow, int k0,
                                               int half) {
  bf16x8 a0 = *(const bf16x8*)(Arow + k0 + 8 * half);
  bf16x8 a1 = *(const bf16x8*)(Arow + k0 + 16 + 8 * half);
  return __builtin_shufflevector(a0, a1, 0, 1, 2, 3, 4, 5, 6, 7,
                                 8, 9, 10, 11, 12, 13, 14, 15);
}
__device__ __forceinline__ bf16x16 load_b_frag(const bf16_t* Wrow, int k0,
                                               int half) {
  bf16x8 b0 = *(const bf16x8*)(Wrow + k0 + 8 * half);
  bf16x8 b1 = *(const bf16x8*)(Wrow + k0 + 16 + 8 * half);
  return __builtin_shufflevector(b0, b1, 0, 1, 2, 3, 4, 5, 6, 7,
                                 8, 9, 10, 11, 12, 13, 14, 15);
}

// ---------------------------------------------------------------------------
// FAST interior GEMM: C[M,N] = A[M,K] @ W[N,K]^T (bf16 in, f32 out).
// Each wave computes a 32(M) x 64(N) tile: 2 A frags x 4 B frags -> 8 WMMA
// per K-step, no guards. Block = 128 threads = 4 waves as 2(M) x 2(N);
// block tile = 64 x 128. Requires: M % 64 == 0, N % 128 == 0, K % 32 == 0.
// ---------------------------------------------------------------------------
__global__ __launch_bounds__(128)
void wmma_gemm_bf16_fast(const bf16_t* __restrict__ A,
                         const bf16_t* __restrict__ W,
                         float* __restrict__ C,
                         int M, int N, int K, int ldc) {
  const int wid  = threadIdx.x >> 5;
  const int lane = threadIdx.x & 31;
  const int half = lane >> 4;
  const int l15  = lane & 15;
  const int wm   = wid & 1;
  const int wn   = wid >> 1;
  const int mbase = blockIdx.y * 64 + wm * 32;   // 2 M-subtiles: +0, +16
  const int nbase = blockIdx.x * 128 + wn * 64;  // 4 N-subtiles: +0,16,32,48

  const bf16_t* Arow[2];
  Arow[0] = A + (long)(mbase + l15) * K;
  Arow[1] = A + (long)(mbase + 16 + l15) * K;
  const bf16_t* Wrow[4];
#pragma unroll
  for (int j = 0; j < 4; ++j)
    Wrow[j] = W + (long)(nbase + j * 16 + l15) * K;

  f32x8 c[2][4];
#pragma unroll
  for (int i = 0; i < 2; ++i)
#pragma unroll
    for (int j = 0; j < 4; ++j)
#pragma unroll
      for (int r = 0; r < 8; ++r) c[i][j][r] = 0.f;

  for (int k0 = 0; k0 < K; k0 += 32) {
    bf16x16 a[2];
#pragma unroll
    for (int i = 0; i < 2; ++i) a[i] = load_a_frag(Arow[i], k0, half);
    bf16x16 b[4];
#pragma unroll
    for (int j = 0; j < 4; ++j) b[j] = load_b_frag(Wrow[j], k0, half);
#pragma unroll
    for (int i = 0; i < 2; ++i)
#pragma unroll
      for (int j = 0; j < 4; ++j)
        c[i][j] = __builtin_amdgcn_wmma_f32_16x16x32_bf16(
            false, a[i], false, b[j], (short)0, c[i][j], false, false);
  }

#pragma unroll
  for (int i = 0; i < 2; ++i) {
#pragma unroll
    for (int j = 0; j < 4; ++j) {
      int n = nbase + j * 16 + l15;
#pragma unroll
      for (int r = 0; r < 8; ++r) {
        int mo = mbase + i * 16 + r + 8 * half;
        C[(long)mo * ldc + n] = c[i][j][r];
      }
    }
  }
}

// ---------------------------------------------------------------------------
// Generic (edge) GEMM: 1 wave per 16x16 tile, guarded on N. Used only for the
// N tail (< 128 columns). All wave-level exits are wave-uniform so EXEC is
// all-ones at the WMMA.
// ---------------------------------------------------------------------------
__global__ __launch_bounds__(128)
void wmma_gemm_bf16_edge(const bf16_t* __restrict__ A,
                         const bf16_t* __restrict__ W,
                         float* __restrict__ C,
                         int M, int N, int K, int ldc) {
  const int wave = threadIdx.x >> 5;
  const int lane = threadIdx.x & 31;
  const int half = lane >> 4;
  const int l15  = lane & 15;
  const int ntile = (blockIdx.x * 4 + wave) * 16;
  const int mtile = blockIdx.y * 16;
  if (ntile >= N || mtile >= M) return;   // wave-uniform

  const int m = mtile + l15;
  const int n = ntile + l15;
  const bf16_t* Arow = A + (long)m * K;
  const bf16_t* Wrow = W + (long)n * K;   // only dereferenced when n < N
  const bool nok = (n < N);

  f32x8 c;
#pragma unroll
  for (int r = 0; r < 8; ++r) c[r] = 0.f;

  for (int k0 = 0; k0 < K; k0 += 32) {
    bf16x16 a = load_a_frag(Arow, k0, half);
    bf16x16 b;
    if (nok) {
      b = load_b_frag(Wrow, k0, half);
    } else {
#pragma unroll
      for (int i = 0; i < 16; ++i) b[i] = (__bf16)0.f;
    }
    c = __builtin_amdgcn_wmma_f32_16x16x32_bf16(
        false, a, false, b, (short)0, c, false, false);
  }

  if (nok) {
#pragma unroll
    for (int r = 0; r < 8; ++r) {
      int mo = mtile + r + 8 * half;
      C[(long)mo * ldc + n] = c[r];
    }
  }
}

// ---------------------------------------------------------------------------
// Causal depthwise conv (width 4, left pad 3) + bias + SiLU over CONV_DIM chans
// ---------------------------------------------------------------------------
__global__ __launch_bounds__(256)
void conv_silu_kernel(const float* __restrict__ zxbcdt,
                      const float* __restrict__ cw,
                      const float* __restrict__ cb,
                      float* __restrict__ xBC) {
  int t = blockIdx.x;
  for (int c = threadIdx.x; c < CONV_DIM; c += 256) {
    float acc = cb[c];
#pragma unroll
    for (int j = 0; j < D_CONV; ++j) {
      int tt = t - (D_CONV - 1) + j;
      if (tt >= 0)
        acc += cw[c * D_CONV + j] *
               zxbcdt[(long)tt * D_IN_PROJ + D_INNER + c];
    }
    float sig = 1.f / (1.f + expf(-acc));
    xBC[(long)t * CONV_DIM + c] = acc * sig;
  }
}

// ---------------------------------------------------------------------------
// dt = softplus(dt_raw + dt_bias); dA = exp(-exp(A_log)*dt)
// ---------------------------------------------------------------------------
__global__ void dt_kernel(const float* __restrict__ zxbcdt,
                          const float* __restrict__ dt_bias,
                          const float* __restrict__ A_log,
                          float* __restrict__ dtb,
                          float* __restrict__ dAb) {
  int i = blockIdx.x * blockDim.x + threadIdx.x;
  if (i >= SEQ_L * NHEADS) return;
  int t = i / NHEADS, h = i % NHEADS;
  float xx = zxbcdt[(long)t * D_IN_PROJ + D_INNER + CONV_DIM + h] + dt_bias[h];
  float sp = (xx > 20.f) ? xx : log1pf(expf(xx));
  dtb[i] = sp;
  dAb[i] = expf(-expf(A_log[h]) * sp);
}

// ---------------------------------------------------------------------------
// Sequential selective-state scan. One block per head; thread p keeps
// h[p][0..63] in registers; B/C rows staged in LDS each step.
// y[t,h,p] = sum_n h[n]*C[n] + Dp[h]*x[t,h,p]
// ---------------------------------------------------------------------------
__global__ __launch_bounds__(64)
void scan_kernel(const float* __restrict__ xBC,
                 const float* __restrict__ dtb,
                 const float* __restrict__ dAb,
                 const float* __restrict__ Dp,
                 float* __restrict__ y) {
  int h = blockIdx.x;
  int p = threadIdx.x;                       // 0..63 (== HEADDIM == D_STATE)
  __shared__ float sB[D_STATE];
  __shared__ float sC[D_STATE];
  float hs[D_STATE];
#pragma unroll
  for (int n = 0; n < D_STATE; ++n) hs[n] = 0.f;
  float Dph = Dp[h];
  for (int t = 0; t < SEQ_L; ++t) {
    const float* row = xBC + (long)t * CONV_DIM;
    sB[p] = row[D_INNER + p];
    sC[p] = row[D_INNER + D_STATE + p];
    __syncthreads();
    float dA  = dAb[t * NHEADS + h];
    float dtv = dtb[t * NHEADS + h];
    float xv  = row[h * HEADDIM + p];
    float dtx = dtv * xv;
    float acc = 0.f;
#pragma unroll
    for (int n = 0; n < D_STATE; ++n) {
      hs[n] = dA * hs[n] + dtx * sB[n];
      acc += hs[n] * sC[n];
    }
    y[(long)t * D_INNER + h * HEADDIM + p] = acc + Dph * xv;
    __syncthreads();
  }
}

// ---------------------------------------------------------------------------
// y *= silu(z); RMSNorm(y) * rms_w -> bf16. One block per row.
// ---------------------------------------------------------------------------
__global__ __launch_bounds__(256)
void gate_rms_kernel(const float* __restrict__ zxbcdt,
                     const float* __restrict__ rms_w,
                     float* __restrict__ y,
                     bf16_t* __restrict__ yn) {
  __shared__ float red[256];
  int t = blockIdx.x;
  float* yr = y + (long)t * D_INNER;
  const float* zr = zxbcdt + (long)t * D_IN_PROJ;
  float ss = 0.f;
  for (int d = threadIdx.x; d < D_INNER; d += 256) {
    float z = zr[d];
    float g = z / (1.f + expf(-z));
    float v = yr[d] * g;
    yr[d] = v;
    ss += v * v;
  }
  red[threadIdx.x] = ss; __syncthreads();
  for (int o = 128; o > 0; o >>= 1) {
    if (threadIdx.x < o) red[threadIdx.x] += red[threadIdx.x + o];
    __syncthreads();
  }
  float rstd = rsqrtf(red[0] / (float)D_INNER + EPS_F);
  for (int d = threadIdx.x; d < D_INNER; d += 256)
    yn[(long)t * D_INNER + d] = (__bf16)(yr[d] * rstd * rms_w[d]);
}

// ---------------------------------------------------------------------------
// x += d
// ---------------------------------------------------------------------------
__global__ void add_kernel(float* __restrict__ x,
                           const float* __restrict__ d, long n) {
  long i = (long)blockIdx.x * blockDim.x + threadIdx.x;
  long stride = (long)gridDim.x * blockDim.x;
  for (; i < n; i += stride) x[i] += d[i];
}

// ---------------------------------------------------------------------------
// Per-row cross-entropy term: partial[t] = (logsumexp(row) - row[target])/L
// ---------------------------------------------------------------------------
__global__ __launch_bounds__(256)
void loss_row_kernel(const float* __restrict__ logits,
                     const int* __restrict__ targets,
                     float* __restrict__ partial) {
  __shared__ float red[256];
  int t = blockIdx.x;
  const float* row = logits + (long)t * VOCAB;
  float m = -1e30f;
  for (int i = threadIdx.x; i < VOCAB; i += 256) m = fmaxf(m, row[i]);
  red[threadIdx.x] = m; __syncthreads();
  for (int o = 128; o > 0; o >>= 1) {
    if (threadIdx.x < o)
      red[threadIdx.x] = fmaxf(red[threadIdx.x], red[threadIdx.x + o]);
    __syncthreads();
  }
  m = red[0]; __syncthreads();
  float s = 0.f;
  for (int i = threadIdx.x; i < VOCAB; i += 256) s += expf(row[i] - m);
  red[threadIdx.x] = s; __syncthreads();
  for (int o = 128; o > 0; o >>= 1) {
    if (threadIdx.x < o) red[threadIdx.x] += red[threadIdx.x + o];
    __syncthreads();
  }
  if (threadIdx.x == 0) {
    float lse = m + logf(red[0]);
    partial[t] = (lse - row[targets[t]]) / (float)SEQ_L;
  }
}

__global__ __launch_bounds__(256)
void loss_final_kernel(const float* __restrict__ partial,
                       float* __restrict__ out) {
  __shared__ float red[256];
  float s = 0.f;
  for (int i = threadIdx.x; i < SEQ_L; i += 256) s += partial[i];
  red[threadIdx.x] = s; __syncthreads();
  for (int o = 128; o > 0; o >>= 1) {
    if (threadIdx.x < o) red[threadIdx.x] += red[threadIdx.x + o];
    __syncthreads();
  }
  if (threadIdx.x == 0) out[0] = red[0];
}

// ---------------------------------------------------------------------------
// Host launcher
// ---------------------------------------------------------------------------
extern "C" void kernel_launch(void* const* d_in, const int* in_sizes, int n_in,
                              void* d_out, int out_size, void* d_ws,
                              size_t ws_size, hipStream_t stream) {
  const int*   idx     = (const int*)d_in[0];
  const int*   targets = (const int*)d_in[1];
  const float* emb     = (const float*)d_in[2];
  const float* ln_w    = (const float*)d_in[3];
  const float* ln_b    = (const float*)d_in[4];
  const float* W_in    = (const float*)d_in[5];
  const float* conv_w  = (const float*)d_in[6];
  const float* conv_b  = (const float*)d_in[7];
  const float* dt_bias = (const float*)d_in[8];
  const float* A_log   = (const float*)d_in[9];
  const float* Dp      = (const float*)d_in[10];
  const float* rms_w   = (const float*)d_in[11];
  const float* W_out   = (const float*)d_in[12];
  const float* lnf_w   = (const float*)d_in[13];
  const float* lnf_b   = (const float*)d_in[14];
  float* logits = (float*)d_out;
  float* loss   = logits + (size_t)SEQ_L * VOCAB;

  // ---- workspace carve-out (256B aligned) ----
  char* p = (char*)d_ws;
  auto carve = [&](size_t bytes) -> void* {
    void* r = (void*)p;
    p += (bytes + 255) & ~(size_t)255;
    return r;
  };
  float*  x     = (float*)carve((size_t)SEQ_L * D_MODEL * 4);
  bf16_t* xnb   = (bf16_t*)carve((size_t)SEQ_L * D_MODEL * 2);
  float*  zx    = (float*)carve((size_t)SEQ_L * D_IN_PROJ * 4);
  float*  xBC   = (float*)carve((size_t)SEQ_L * CONV_DIM * 4);
  float*  dtb   = (float*)carve((size_t)SEQ_L * NHEADS * 4);
  float*  dAb   = (float*)carve((size_t)SEQ_L * NHEADS * 4);
  float*  ybuf  = (float*)carve((size_t)SEQ_L * D_INNER * 4);
  bf16_t* ynb   = (bf16_t*)carve((size_t)SEQ_L * D_INNER * 2);
  float*  otmp  = (float*)carve((size_t)SEQ_L * D_MODEL * 4);
  bf16_t* Winb  = (bf16_t*)carve((size_t)N_LAYER * D_IN_PROJ * D_MODEL * 2);
  bf16_t* Woutb = (bf16_t*)carve((size_t)N_LAYER * D_MODEL * D_INNER * 2);
  bf16_t* embb  = (bf16_t*)carve((size_t)VOCAB * D_MODEL * 2);
  float*  part  = (float*)carve((size_t)SEQ_L * 4);

  // GEMM dispatcher: fast interior kernel + guarded tail kernel.
  auto launch_gemm = [&](const bf16_t* A, const bf16_t* W, float* C,
                         int M, int N, int K, int ldc) {
    int Nfast = N & ~127;            // multiple of 128
    if (Nfast > 0) {
      dim3 grid(Nfast / 128, M / 64);
      wmma_gemm_bf16_fast<<<grid, 128, 0, stream>>>(A, W, C, M, Nfast, K, ldc);
    }
    int tail = N - Nfast;
    if (tail > 0) {
      dim3 grid((tail + 63) / 64, M / 16);
      wmma_gemm_bf16_edge<<<grid, 128, 0, stream>>>(
          A, W + (size_t)Nfast * K, C + Nfast, M, tail, K, ldc);
    }
  };

  // ---- weight conversions to bf16 (every call; deterministic) ----
  {
    long n1 = (long)N_LAYER * D_IN_PROJ * D_MODEL;
    f32_to_bf16_kernel<<<2048, 256, 0, stream>>>(W_in, Winb, n1);
    long n2 = (long)N_LAYER * D_MODEL * D_INNER;
    f32_to_bf16_kernel<<<2048, 256, 0, stream>>>(W_out, Woutb, n2);
    long n3 = (long)VOCAB * D_MODEL;
    f32_to_bf16_kernel<<<4096, 256, 0, stream>>>(emb, embb, n3);
  }

  // ---- embed ----
  embed_kernel<<<SEQ_L, 256, 0, stream>>>(idx, emb, x);

  // ---- layers ----
  for (int l = 0; l < N_LAYER; ++l) {
    layernorm_bf16_kernel<<<SEQ_L, 256, 0, stream>>>(
        x, ln_w + l * D_MODEL, ln_b + l * D_MODEL, xnb, D_MODEL);

    // zxbcdt = xn @ W_in^T  (1024 x 3224)
    launch_gemm(xnb, Winb + (size_t)l * D_IN_PROJ * D_MODEL, zx,
                SEQ_L, D_IN_PROJ, D_MODEL, D_IN_PROJ);

    conv_silu_kernel<<<SEQ_L, 256, 0, stream>>>(
        zx, conv_w + (size_t)l * CONV_DIM * D_CONV, conv_b + l * CONV_DIM, xBC);

    dt_kernel<<<(SEQ_L * NHEADS + 255) / 256, 256, 0, stream>>>(
        zx, dt_bias + l * NHEADS, A_log + l * NHEADS, dtb, dAb);

    scan_kernel<<<NHEADS, 64, 0, stream>>>(xBC, dtb, dAb, Dp + l * NHEADS, ybuf);

    gate_rms_kernel<<<SEQ_L, 256, 0, stream>>>(zx, rms_w + l * D_INNER, ybuf, ynb);

    // otmp = yn @ W_out^T  (1024 x 768)
    launch_gemm(ynb, Woutb + (size_t)l * D_MODEL * D_INNER, otmp,
                SEQ_L, D_MODEL, D_INNER, D_MODEL);

    add_kernel<<<1024, 256, 0, stream>>>(x, otmp, (long)SEQ_L * D_MODEL);
  }

  // ---- final LN + logits GEMM into d_out ----
  layernorm_bf16_kernel<<<SEQ_L, 256, 0, stream>>>(x, lnf_w, lnf_b, xnb, D_MODEL);
  launch_gemm(xnb, embb, logits, SEQ_L, VOCAB, D_MODEL, VOCAB);

  // ---- loss ----
  loss_row_kernel<<<SEQ_L, 256, 0, stream>>>(logits, targets, part);
  loss_final_kernel<<<1, 256, 0, stream>>>(part, loss);
}
